// GAPCN_48945447305282
// MI455X (gfx1250) — matrix operand order repaired and checked by reading, not compiled, not run on hardware
//
#include <hip/hip_runtime.h>

// ---------------------------------------------------------------------------
// MI455X (gfx1250) implementation. wave32, WMMA f16 (f32 accumulate).
// All activations stored position-major f16 with channel stride padded to a
// multiple of 32 so every WMMA fragment load is a contiguous 16B/32B load.
// BN (scale/bias) + LeakyReLU(0.2) fused into the GEMM epilogue.
// ---------------------------------------------------------------------------

typedef __attribute__((ext_vector_type(16))) _Float16 v16h;
typedef __attribute__((ext_vector_type(8)))  _Float16 v8h;
typedef __attribute__((ext_vector_type(8)))  float    v8f;

constexpr int Bb  = 2;
constexpr int Nn  = 2048;
constexpr int KK  = 20;
constexpr int Pp  = Bb * Nn;        // 4096 points
constexpr int PKt = Pp * KK;        // 81920 (point, neighbor) pairs

__device__ __forceinline__ float lrelu(float v) { return v > 0.f ? v : 0.2f * v; }

// ---------------------------------------------------------------------------
// KNN: one thread per point, maintain top-20 of pd = -||pi - pj||^2
// (matches jax top_k: descending, stable/earliest index on ties).
// Also emits dis_k = 2*(1 - sigmoid(pd)).
// ---------------------------------------------------------------------------
__global__ void knn_kernel(const float* __restrict__ x, int* __restrict__ idx,
                           float* __restrict__ disk) {
  int p = blockIdx.x * blockDim.x + threadIdx.x;
  if (p >= Pp) return;
  int b = p / Nn, n = p % Nn;
  const float* pos = x + (size_t)b * 4 * Nn;
  float px = pos[n], py = pos[Nn + n], pz = pos[2 * Nn + n];
  float xxn = px * px + py * py + pz * pz;
  float best[KK];
  int   bidx[KK];
#pragma unroll
  for (int k = 0; k < KK; ++k) { best[k] = -3.4e38f; bidx[k] = 0; }
  for (int m = 0; m < Nn; ++m) {
    float qx = pos[m], qy = pos[Nn + m], qz = pos[2 * Nn + m];
    float pd = 2.f * (px * qx + py * qy + pz * qz) - xxn
             - (qx * qx + qy * qy + qz * qz);
    if (pd > best[KK - 1]) {
      int j = KK - 1;
      while (j > 0 && best[j - 1] < pd) {
        best[j] = best[j - 1]; bidx[j] = bidx[j - 1]; --j;
      }
      best[j] = pd; bidx[j] = m;
    }
  }
  for (int k = 0; k < KK; ++k) {
    idx[(size_t)p * KK + k] = bidx[k];
    float sg = 1.f / (1.f + expf(-best[k]));
    disk[(size_t)p * KK + k] = 2.f * (1.f - sg);
  }
}

// ---------------------------------------------------------------------------
// PCA normals: smallest eigenvector of 3x3 neighbor covariance (analytic).
// ---------------------------------------------------------------------------
__global__ void normals_kernel(const float* __restrict__ x,
                               const int* __restrict__ idx,
                               float* __restrict__ nrm) {
  int p = blockIdx.x * blockDim.x + threadIdx.x;
  if (p >= Pp) return;
  int b = p / Nn;
  const float* pos = x + (size_t)b * 4 * Nn;
  float mx = 0.f, my = 0.f, mz = 0.f;
  for (int k = 0; k < KK; ++k) {
    int m = idx[(size_t)p * KK + k];
    mx += pos[m]; my += pos[Nn + m]; mz += pos[2 * Nn + m];
  }
  const float inv = 1.f / (float)KK;
  mx *= inv; my *= inv; mz *= inv;
  float a00 = 0, a01 = 0, a02 = 0, a11 = 0, a12 = 0, a22 = 0;
  for (int k = 0; k < KK; ++k) {
    int m = idx[(size_t)p * KK + k];
    float cx = pos[m] - mx, cy = pos[Nn + m] - my, cz = pos[2 * Nn + m] - mz;
    a00 += cx * cx; a01 += cx * cy; a02 += cx * cz;
    a11 += cy * cy; a12 += cy * cz; a22 += cz * cz;
  }
  float q = (a00 + a11 + a22) / 3.f;
  float b00 = a00 - q, b11 = a11 - q, b22 = a22 - q;
  float p2 = a01 * a01 + a02 * a02 + a12 * a12;
  float p1 = b00 * b00 + b11 * b11 + b22 * b22 + 2.f * p2;
  float pp = sqrtf(p1 / 6.f) + 1e-20f;
  float ip = 1.f / pp;
  float c00 = b00 * ip, c11 = b11 * ip, c22 = b22 * ip;
  float c01 = a01 * ip, c02 = a02 * ip, c12 = a12 * ip;
  float detB = c00 * (c11 * c22 - c12 * c12) - c01 * (c01 * c22 - c12 * c02)
             + c02 * (c01 * c12 - c11 * c02);
  float r = 0.5f * detB;
  r = fminf(1.f, fmaxf(-1.f, r));
  float phi = acosf(r) / 3.f;
  float lmin = q + 2.f * pp * cosf(phi + 2.0943951023931953f);
  float m00 = a00 - lmin, m11 = a11 - lmin, m22 = a22 - lmin;
  // cross products of row pairs of (A - lmin I)
  float v0x = a01 * a12 - a02 * m11, v0y = a02 * a01 - m00 * a12, v0z = m00 * m11 - a01 * a01;
  float v1x = a01 * m22 - a12 * a02, v1y = a02 * a02 - m00 * m22, v1z = m00 * a12 - a01 * a02;
  float v2x = m11 * m22 - a12 * a12, v2y = a12 * a02 - a01 * m22, v2z = a01 * a12 - m11 * a02;
  float n0 = v0x * v0x + v0y * v0y + v0z * v0z;
  float n1 = v1x * v1x + v1y * v1y + v1z * v1z;
  float n2 = v2x * v2x + v2y * v2y + v2z * v2z;
  float vx = v0x, vy = v0y, vz = v0z, nb = n0;
  if (n1 > nb) { vx = v1x; vy = v1y; vz = v1z; nb = n1; }
  if (n2 > nb) { vx = v2x; vy = v2y; vz = v2z; nb = n2; }
  float nn = sqrtf(nb);
  if (nn < 1e-12f) { vx = 0.f; vy = 0.f; vz = 1.f; nn = 1.f; }
  float in2 = 1.f / nn;
  nrm[(size_t)p * 4 + 0] = vx * in2;
  nrm[(size_t)p * 4 + 1] = vy * in2;
  nrm[(size_t)p * 4 + 2] = vz * in2;
  nrm[(size_t)p * 4 + 3] = 0.f;
}

// ---------------------------------------------------------------------------
// Weight pack: fp32 [Cout][Cin] -> f16 [Cout][CinPad], zero padded.
// ---------------------------------------------------------------------------
__global__ void pack_w_kernel(const float* __restrict__ w, int cout, int cin,
                              _Float16* __restrict__ dst, int cinp) {
  int t = blockIdx.x * blockDim.x + threadIdx.x;
  int total = cout * cinp;
  if (t >= total) return;
  int ci = t % cinp, co = t / cinp;
  dst[t] = (_Float16)(ci < cin ? w[(size_t)co * cin + ci] : 0.f);
}

// ---------------------------------------------------------------------------
// Edge-mix: dst[pk][c] = nb-ctr (c<C), ctr (C<=c<2C), else 0. f16 in/out.
// ---------------------------------------------------------------------------
__global__ void edge_mix_kernel(const _Float16* __restrict__ src, int lds_, int C,
                                const int* __restrict__ idx,
                                _Float16* __restrict__ dst, int Dld, int total) {
  int t = blockIdx.x * blockDim.x + threadIdx.x;
  if (t >= total) return;
  int c = t % Dld, pk = t / Dld;
  int p = pk / KK, b = p / Nn;
  int pnb = b * Nn + idx[pk];
  _Float16 o = (_Float16)0.f;
  if (c < C)          o = src[(size_t)pnb * lds_ + c] - src[(size_t)p * lds_ + c];
  else if (c < 2 * C) o = src[(size_t)p * lds_ + (c - C)];
  dst[(size_t)pk * Dld + c] = o;
}

// Variant with dis_k scaling of the 2C edge channels + 3 normals-edge channels.
__global__ void edge_mix_dn_kernel(const _Float16* __restrict__ src, int lds_, int C,
                                   const int* __restrict__ idx,
                                   const float* __restrict__ disk,
                                   const float* __restrict__ nrm,
                                   _Float16* __restrict__ dst, int Dld, int total) {
  int t = blockIdx.x * blockDim.x + threadIdx.x;
  if (t >= total) return;
  int c = t % Dld, pk = t / Dld;
  int p = pk / KK, b = p / Nn;
  int pnb = b * Nn + idx[pk];
  _Float16 o = (_Float16)0.f;
  if (c < 2 * C) {
    float v = (c < C)
      ? ((float)src[(size_t)pnb * lds_ + c] - (float)src[(size_t)p * lds_ + c])
      : (float)src[(size_t)p * lds_ + (c - C)];
    o = (_Float16)(v * disk[pk]);
  } else if (c < 2 * C + 3) {
    int cc = c - 2 * C;
    o = (_Float16)(nrm[(size_t)pnb * 4 + cc] - nrm[(size_t)p * 4 + cc]);
  }
  dst[(size_t)pk * Dld + c] = o;
}

// ---------------------------------------------------------------------------
// WMMA GEMM + BN + LeakyReLU.
//   Y[p, coff+co] = act( s[co] * sum_ci X[p,ci]*Wp[co,ci] + b[co] )
// X: [Pn][ldx] f16 (ldx multiple of 32), Wp: [cout][cinp] f16 zero-padded.
// Block: 256 thr = 8 waves; each wave: 32 positions x 16 couts
// (two v_wmma_f32_16x16x32_f16 accumulators, K-loop over cinp).
// A-fragment (16-bit A 16x32): lane holds row M=lane&15,
//   K in [kg*16, kg*16+16) (two 8-wide loads), kg = lane>>4.
// B-fragment (32x16): lane holds column co=lane&15, K in [kg*16, kg*16+16)
//   -> contiguous in W[co][*] row-major.
// C/D: vgpr v -> row v + kg*8, col lane&15.
// ---------------------------------------------------------------------------
__global__ __launch_bounds__(256) void gemm_bn_act_kernel(
    const _Float16* __restrict__ X, int ldx,
    const _Float16* __restrict__ Wp, int cinp, int cout,
    const float* __restrict__ sc, const float* __restrict__ bi,
    int act, _Float16* __restrict__ Y, int ldy, int coff) {
  int tid  = threadIdx.x;
  int lane = tid & 31, w = tid >> 5;
  int pw = w & 3, cw = w >> 2;
  int l15 = lane & 15, kg = lane >> 4;
  int prow = blockIdx.x * 128 + pw * 32;
  int cob  = blockIdx.y * 32 + cw * 16;
  if (cob >= cout) return;  // wave-uniform

  const _Float16* ap = X  + (size_t)(prow + l15) * ldx + kg * 16;
  const _Float16* bp = Wp + (size_t)(cob  + l15) * cinp + kg * 16;
  const size_t arow16 = (size_t)16 * ldx;

  v8f acc0 = {0.f, 0.f, 0.f, 0.f, 0.f, 0.f, 0.f, 0.f};
  v8f acc1 = acc0;

  for (int ci = 0; ci < cinp; ci += 32) {
    v8h a0lo = *(const v8h*)(ap + ci);
    v8h a0hi = *(const v8h*)(ap + ci + 8);
    v8h a1lo = *(const v8h*)(ap + ci + arow16);
    v8h a1hi = *(const v8h*)(ap + ci + 8 + arow16);
    v16h bb  = *(const v16h*)(bp + ci);
    __builtin_prefetch(ap + ci + 128, 0, 0);   // global_prefetch_b8
    v16h a0, a1;
#pragma unroll
    for (int i = 0; i < 8; ++i) {
      a0[i] = a0lo[i]; a0[i + 8] = a0hi[i];
      a1[i] = a1lo[i]; a1[i + 8] = a1hi[i];
    }
    acc0 = __builtin_amdgcn_wmma_f32_16x16x32_f16(false, a0, false, bb,
                                                  (short)0, acc0, false, false);
    acc1 = __builtin_amdgcn_wmma_f32_16x16x32_f16(false, a1, false, bb,
                                                  (short)0, acc1, false, false);
  }

  int co = cob + l15;
  float s = sc ? sc[co] : 1.f;
  float b = bi ? bi[co] : 0.f;
  size_t y0 = (size_t)(prow + kg * 8)      * ldy + coff + co;
  size_t y1 = (size_t)(prow + 16 + kg * 8) * ldy + coff + co;
#pragma unroll
  for (int v = 0; v < 8; ++v) {
    float r0 = acc0[v] * s + b; if (act) r0 = lrelu(r0);
    float r1 = acc1[v] * s + b; if (act) r1 = lrelu(r1);
    Y[y0 + (size_t)v * ldy] = (_Float16)r0;
    Y[y1 + (size_t)v * ldy] = (_Float16)r1;
  }
}

// ---------------------------------------------------------------------------
// Attention pieces (Cout==1 convs use a plain dot, WMMA would waste 15/16).
// t[pk] = lrelu( s_a*(Wa.h) + b_a + s_e*(We.eo) + b_e )
// ---------------------------------------------------------------------------
__global__ void dot_ae_kernel(const _Float16* __restrict__ H, int ldh,
                              const float* __restrict__ Wa, const float* __restrict__ sa,
                              const float* __restrict__ ba,
                              const _Float16* __restrict__ E, int lde,
                              const float* __restrict__ We, const float* __restrict__ se,
                              const float* __restrict__ be,
                              int cin, float* __restrict__ t) {
  int pk = blockIdx.x * blockDim.x + threadIdx.x;
  if (pk >= PKt) return;
  float a = 0.f, e = 0.f;
  for (int ci = 0; ci < cin; ++ci) {
    a += (float)H[(size_t)pk * ldh + ci] * Wa[ci];
    e += (float)E[(size_t)pk * lde + ci] * We[ci];
  }
  float v = (sa[0] * a + ba[0]) + (se[0] * e + be[0]);
  t[pk] = lrelu(v);
}

__global__ void softmax_k_kernel(float* __restrict__ t) {
  int p = blockIdx.x * blockDim.x + threadIdx.x;
  if (p >= Pp) return;
  float m = -3.4e38f;
  for (int k = 0; k < KK; ++k) m = fmaxf(m, t[(size_t)p * KK + k]);
  float s = 0.f;
  for (int k = 0; k < KK; ++k) s += expf(t[(size_t)p * KK + k] - m);
  float is = 1.f / s;
  for (int k = 0; k < KK; ++k)
    t[(size_t)p * KK + k] = expf(t[(size_t)p * KK + k] - m) * is;
}

// out[p, ooff+c] = sum_k att[p,k] * eo[p,k,c]
__global__ void attn_apply_kernel(const float* __restrict__ att,
                                  const _Float16* __restrict__ eo, int lde, int C,
                                  _Float16* __restrict__ outb, int ldo, int ooff,
                                  int total) {
  int t = blockIdx.x * blockDim.x + threadIdx.x;
  if (t >= total) return;
  int c = t % C, p = t / C;
  float s = 0.f;
  for (int k = 0; k < KK; ++k)
    s += att[(size_t)p * KK + k] * (float)eo[((size_t)p * KK + k) * lde + c];
  outb[(size_t)p * ldo + ooff + c] = (_Float16)s;
}

// out[p, ooff+c] = max_k in[p,k,c]
__global__ void maxk_kernel(const _Float16* __restrict__ in, int ldin, int C,
                            _Float16* __restrict__ outb, int ldo, int ooff,
                            int total) {
  int t = blockIdx.x * blockDim.x + threadIdx.x;
  if (t >= total) return;
  int c = t % C, p = t / C;
  float m = -3.4e38f;
  for (int k = 0; k < KK; ++k)
    m = fmaxf(m, (float)in[((size_t)p * KK + k) * ldin + c]);
  outb[(size_t)p * ldo + ooff + c] = (_Float16)m;
}

__global__ void copyh_kernel(const _Float16* __restrict__ src, int lds_, int C,
                             _Float16* __restrict__ dst, int ldd, int doff,
                             int total) {
  int t = blockIdx.x * blockDim.x + threadIdx.x;
  if (t >= total) return;
  int c = t % C, p = t / C;
  dst[(size_t)p * ldd + doff + c] = src[(size_t)p * lds_ + c];
}

// featH[p][0] = feat channel, recf[p] = feat (fp32 residual)
__global__ void prep_kernel(const float* __restrict__ x,
                            _Float16* __restrict__ featH, float* __restrict__ recf) {
  int p = blockIdx.x * blockDim.x + threadIdx.x;
  if (p >= Pp) return;
  int b = p / Nn, n = p % Nn;
  float f = x[(size_t)b * 4 * Nn + 3 * Nn + n];
  featH[(size_t)p * 32] = (_Float16)f;
  recf[p] = f;
}

// xcf[p] = [rec(1), x1(64), pad..96]
__global__ void xcfeat_kernel(const float* __restrict__ recf,
                              const _Float16* __restrict__ x1b,
                              _Float16* __restrict__ dst) {
  int t = blockIdx.x * blockDim.x + threadIdx.x;
  int total = Pp * 96;
  if (t >= total) return;
  int c = t % 96, p = t / 96;
  _Float16 o = (_Float16)0.f;
  if (c == 0)      o = (_Float16)recf[p];
  else if (c < 65) o = x1b[(size_t)p * 64 + (c - 1)];
  dst[(size_t)p * 96 + c] = o;
}

// Final 128->1 conv (s=1,b=0 in reference) + residual; fp32 output.
__global__ void final_kernel(const _Float16* __restrict__ q2,
                             const float* __restrict__ W3,
                             const float* __restrict__ s3,
                             const float* __restrict__ b3,
                             const float* __restrict__ recf,
                             float* __restrict__ out) {
  int p = blockIdx.x * blockDim.x + threadIdx.x;
  if (p >= Pp) return;
  float a = 0.f;
  for (int ci = 0; ci < 128; ++ci)
    a += (float)q2[(size_t)p * 128 + ci] * W3[ci];
  out[p] = (a * s3[0] + b3[0]) + recf[p];
}

// ---------------------------------------------------------------------------
// Host orchestration.
// Param flattening assumed = setup_inputs() dict insertion order, depth-first:
//   d_in[0]=x; gap1{s1,s2,s3 each c1(W,s,b),c2,c1n,c2n}; gap2{...};
//   conv1[3x(W,s,b)]; conv2; conv3; convn2; convn1.   (118 leaves total)
// Workspace arena: ~315 MB (largest: em2 = 81920 x 672 f16 = 110 MB).
// ---------------------------------------------------------------------------
extern "C" void kernel_launch(void* const* d_in, const int* in_sizes, int n_in,
                              void* d_out, int out_size, void* d_ws, size_t ws_size,
                              hipStream_t stream) {
  (void)in_sizes; (void)n_in; (void)out_size; (void)ws_size;
  const float* xin = (const float*)d_in[0];
  auto Wf = [&](int i) -> const float* { return (const float*)d_in[i]; };

  char* wsb = (char*)d_ws;
  size_t off = 0;
  auto alloc = [&](size_t bytes) -> void* {
    off = (off + 255) & ~(size_t)255;
    void* p = wsb + off;
    off += bytes;
    return p;
  };

  int*      idxb  = (int*)     alloc((size_t)PKt * 4);
  float*    diskb = (float*)   alloc((size_t)PKt * 4);
  float*    tb    = (float*)   alloc((size_t)PKt * 4);
  float*    nrm   = (float*)   alloc((size_t)Pp * 4 * 4);
  float*    recf  = (float*)   alloc((size_t)Pp * 4);
  _Float16* featH = (_Float16*)alloc((size_t)Pp * 32 * 2);
  _Float16* e2a   = (_Float16*)alloc((size_t)PKt * 32 * 2);
  _Float16* e2b   = (_Float16*)alloc((size_t)PKt * 64 * 2);
  _Float16* xn1   = (_Float16*)alloc((size_t)PKt * 64 * 2);
  _Float16* x1b   = (_Float16*)alloc((size_t)Pp * 64 * 2);
  _Float16* hb    = (_Float16*)alloc((size_t)PKt * 128 * 2);
  _Float16* xcf   = (_Float16*)alloc((size_t)Pp * 96 * 2);
  _Float16* em1   = (_Float16*)alloc((size_t)PKt * 160 * 2);
  _Float16* t1    = (_Float16*)alloc((size_t)PKt * 128 * 2);
  _Float16* t2    = (_Float16*)alloc((size_t)PKt * 64 * 2);
  _Float16* hfeat = (_Float16*)alloc((size_t)Pp * 64 * 2);
  _Float16* xn2   = (_Float16*)alloc((size_t)PKt * 256 * 2);
  _Float16* x2b   = (_Float16*)alloc((size_t)Pp * 320 * 2);
  _Float16* e2d   = (_Float16*)alloc((size_t)PKt * 256 * 2);
  _Float16* u2    = (_Float16*)alloc((size_t)PKt * 64 * 2);
  _Float16* fb    = (_Float16*)alloc((size_t)Pp * 576 * 2);
  _Float16* em2   = (_Float16*)alloc((size_t)PKt * 672 * 2);

  auto cdiv = [](int a, int b) { return (a + b - 1) / b; };

  auto pack = [&](int widx, int cout, int cin, int cinp) -> const _Float16* {
    _Float16* dst = (_Float16*)alloc((size_t)cout * cinp * 2);
    int total = cout * cinp;
    pack_w_kernel<<<cdiv(total, 256), 256, 0, stream>>>(Wf(widx), cout, cin, dst, cinp);
    return dst;
  };

  auto gemm = [&](const _Float16* X, int ldx, const _Float16* Wp, int cinp,
                  int cout, int Pn, const float* s, const float* b, bool act,
                  _Float16* Y, int ldy, int coff) {
    dim3 g(Pn / 128, cdiv(cout, 32));
    gemm_bn_act_kernel<<<g, 256, 0, stream>>>(X, ldx, Wp, cinp, cout, s, b,
                                              (int)act, Y, ldy, coff);
  };

  auto conv_bn = [&](const _Float16* X, int ldx, int base, int cout, int cin,
                     int cinp, _Float16* Y, int ldy, bool act, int Pn) {
    const _Float16* wp = pack(base, cout, cin, cinp);
    gemm(X, ldx, wp, cinp, cout, Pn, Wf(base + 1), Wf(base + 2), act, Y, ldy, 0);
  };

  auto edge_mix = [&](const _Float16* src, int lds_, int C, _Float16* dst, int Dld) {
    int total = PKt * Dld;
    edge_mix_kernel<<<cdiv(total, 256), 256, 0, stream>>>(src, lds_, C, idxb,
                                                          dst, Dld, total);
  };
  auto edge_mix_dn = [&](const _Float16* src, int lds_, int C, _Float16* dst, int Dld) {
    int total = PKt * Dld;
    edge_mix_dn_kernel<<<cdiv(total, 256), 256, 0, stream>>>(src, lds_, C, idxb,
                                                             diskb, nrm, dst, Dld, total);
  };
  auto maxk = [&](const _Float16* in, int ldin, int C, _Float16* o, int ldo, int ooff) {
    int total = Pp * C;
    maxk_kernel<<<cdiv(total, 256), 256, 0, stream>>>(in, ldin, C, o, ldo, ooff, total);
  };

  // GAP single: c1/c1n GEMMs -> a/e dots -> softmax -> attention-weighted sum.
  auto run_single = [&](const _Float16* e2, int lde2, int cin2, int cinp2, int ch,
                        int base, _Float16* eobuf, int ldeo, int eooff,
                        _Float16* outbuf, int ldo, int outoff) {
    const _Float16* w1  = pack(base + 0, ch, cin2, cinp2);
    const _Float16* w1n = pack(base + 6, ch, cin2, cinp2);
    gemm(e2, lde2, w1,  cinp2, ch, PKt, Wf(base + 1), Wf(base + 2), true, hb, 128, 0);
    gemm(e2, lde2, w1n, cinp2, ch, PKt, Wf(base + 7), Wf(base + 8), true, eobuf, ldeo, eooff);
    dot_ae_kernel<<<cdiv(PKt, 256), 256, 0, stream>>>(
        hb, 128, Wf(base + 3), Wf(base + 4), Wf(base + 5),
        eobuf + eooff, ldeo, Wf(base + 9), Wf(base + 10), Wf(base + 11), ch, tb);
    softmax_k_kernel<<<cdiv(Pp, 256), 256, 0, stream>>>(tb);
    int total = Pp * ch;
    attn_apply_kernel<<<cdiv(total, 256), 256, 0, stream>>>(
        tb, eobuf + eooff, ldeo, ch, outbuf, ldo, outoff, total);
  };

  // ---- stage 0: features / knn / normals ----
  prep_kernel<<<cdiv(Pp, 256), 256, 0, stream>>>(xin, featH, recf);
  knn_kernel<<<cdiv(Pp, 128), 128, 0, stream>>>(xin, idxb, diskb);
  normals_kernel<<<cdiv(Pp, 128), 128, 0, stream>>>(xin, idxb, nrm);

  // ---- gap1 (ch=16, cin=1) ----
  edge_mix(featH, 32, 1, e2a, 32);
  run_single(e2a, 32, 2, 32, 16, 1,  xn1, 64, 0,  x1b, 64, 0);
  run_single(e2a, 32, 2, 32, 16, 13, xn1, 64, 16, x1b, 64, 16);
  edge_mix(x1b, 64, 32, e2b, 64);                       // s3 input: xc = x1b[:,0:32]
  run_single(e2b, 64, 64, 64, 32, 25, xn1, 64, 32, x1b, 64, 32);

  // ---- em = concat(edge_mix(rec|x1)*dis_k, normals_edge) -> conv1 -> max_k ----
  xcfeat_kernel<<<cdiv(Pp * 96, 256), 256, 0, stream>>>(recf, x1b, xcf);
  edge_mix_dn(xcf, 96, 65, em1, 160);                   // 133 ch, pad 160
  conv_bn(em1, 160, 73, 128, 133, 160, t1, 128, true, PKt);
  conv_bn(t1, 128, 76, 64, 128, 128, t2, 64, true, PKt);
  _Float16* t3 = em1;                                   // reuse em1 as [PKt,64]
  conv_bn(t2, 64, 79, 64, 64, 64, t3, 64, true, PKt);
  maxk(t3, 64, 64, hfeat, 64, 0);                       // x_feat

  // ---- gap2 (ch=64, cin=64) ----
  _Float16* e2c = t1;                                   // reuse [PKt,128]
  edge_mix(hfeat, 64, 64, e2c, 128);
  run_single(e2c, 128, 128, 128, 64, 37, xn2, 256, 0,  x2b, 320, 0);
  run_single(e2c, 128, 128, 128, 64, 49, xn2, 256, 64, x2b, 320, 64);
  edge_mix(x2b, 320, 128, e2d, 256);                    // s3 input: xc2 = x2b[:,0:128]
  run_single(e2d, 256, 256, 256, 128, 61, xn2, 256, 128, x2b, 320, 128);

  // ---- convn1(_xn) -> max_k -> f[:,0:64] ----
  _Float16* u1 = t2;                                    // reuse [PKt,64]
  conv_bn(xn1, 64, 109, 64, 64, 64, u1, 64, true, PKt);
  conv_bn(u1, 64, 112, 64, 64, 64, u2, 64, true, PKt);
  conv_bn(u2, 64, 115, 64, 64, 64, u1, 64, true, PKt);
  maxk(u1, 64, 64, fb, 576, 0);

  // ---- convn2(xn2) -> max_k -> f[:,320:576] ----
  conv_bn(xn2, 256, 100, 128, 256, 256, hb, 128, true, PKt);
  conv_bn(hb, 128, 103, 128, 128, 128, e2c, 128, true, PKt);
  conv_bn(e2c, 128, 106, 256, 128, 128, e2d, 256, true, PKt);
  maxk(e2d, 256, 256, fb, 576, 320);

  // ---- x2c = concat(x2, x_feat) ; em2 ; conv3 -> max_k -> f[:,64:320] ----
  copyh_kernel<<<cdiv(Pp * 64, 256), 256, 0, stream>>>(hfeat, 64, 64, x2b, 320, 256,
                                                       Pp * 64);
  edge_mix_dn(x2b, 320, 320, em2, 672);                 // 643 ch, pad 672
  _Float16* g1 = xn2;                                   // reuse [PKt,>=192]
  conv_bn(em2, 672, 91, 192, 643, 672, g1, 192, true, PKt);
  conv_bn(g1, 192, 94, 128, 192, 192, hb, 128, true, PKt);
  conv_bn(hb, 128, 97, 256, 128, 128, e2d, 256, true, PKt);
  maxk(e2d, 256, 256, fb, 576, 64);

  // ---- conv2 head (per point) + residual ----
  _Float16* q1 = x2b;                                   // reuse as [Pp,256]
  conv_bn(fb, 576, 82, 256, 576, 576, q1, 256, true, Pp);
  _Float16* q2 = fb;                                    // reuse as [Pp,128]
  conv_bn(q1, 256, 85, 128, 256, 256, q2, 128, true, Pp);
  final_kernel<<<cdiv(Pp, 256), 256, 0, stream>>>(q2, Wf(88), Wf(89), Wf(90),
                                                  recf, (float*)d_out);
}
